// OnlyDecoder_47210280517805
// MI455X (gfx1250) — matrix operand order repaired and checked by reading, not compiled, not run on hardware
//
#include <hip/hip_runtime.h>
#include <hip/hip_bf16.h>

typedef __attribute__((ext_vector_type(16))) _Float16 v16h;
typedef __attribute__((ext_vector_type(8)))  float    v8f;

#define WMMA_F16(a, b, c) \
  __builtin_amdgcn_wmma_f32_16x16x32_f16(false, (a), false, (b), (short)0, (c), false, false)

// A-layout column permutation: row stored as [k0..7, k16..23, k8..15, k24..31]
// (swap bits 3 and 4 of k). Then lane fragment = 16 contiguous halves at half*16.
__device__ __forceinline__ int aperm(int c) {
  return (c & 7) | ((c & 8) << 1) | ((c & 16) >> 1);
}

// 32-byte LDS fragment load (16-byte aligned) -> 2x ds_load_b128
__device__ __forceinline__ v16h frag_ld(const _Float16* p) {
  v16h r;
  __builtin_memcpy(&r, p, sizeof(v16h));
  return r;
}

// Async global->LDS copy of 48 bytes (3x B128), tracked by ASYNCcnt.
__device__ __forceinline__ void async_copy48(unsigned ldsAddr, const _Float16* src) {
  unsigned long long ga = (unsigned long long)(size_t)src;
  asm volatile("global_load_async_to_lds_b128 %0, %1, off"
               :: "v"(ldsAddr), "v"(ga) : "memory");
  asm volatile("global_load_async_to_lds_b128 %0, %1, off offset:16"
               :: "v"(ldsAddr), "v"(ga) : "memory");
  asm volatile("global_load_async_to_lds_b128 %0, %1, off offset:32"
               :: "v"(ldsAddr), "v"(ga) : "memory");
}
__device__ __forceinline__ void wait_async_keep3() {
  asm volatile("s_wait_asynccnt 0x3" ::: "memory");
}
__device__ __forceinline__ void wait_async_all() {
  asm volatile("s_wait_asynccnt 0x0" ::: "memory");
}

// ---------------------------------------------------------------- utilities
__device__ __forceinline__ float half16_max(float v) {
  v = fmaxf(v, __shfl_xor(v, 1, 32));
  v = fmaxf(v, __shfl_xor(v, 2, 32));
  v = fmaxf(v, __shfl_xor(v, 4, 32));
  v = fmaxf(v, __shfl_xor(v, 8, 32));
  return v;
}
__device__ __forceinline__ float half16_sum(float v) {
  v += __shfl_xor(v, 1, 32);
  v += __shfl_xor(v, 2, 32);
  v += __shfl_xor(v, 4, 32);
  v += __shfl_xor(v, 8, 32);
  return v;
}

// ---------------------------------------------------------------- embedding
__global__ void embed_kernel(const int* __restrict__ idx,
                             const float* __restrict__ tok,
                             const float* __restrict__ pos,
                             float* __restrict__ x, size_t total) {
  const int E = 768, T = 1024;
  size_t i = (size_t)blockIdx.x * blockDim.x + threadIdx.x;
  if (i >= total) return;
  int e = (int)(i % E);
  size_t bt = i / E;
  int t = (int)(bt % T);
  x[i] = tok[(size_t)idx[bt] * E + e] + pos[(size_t)t * E + e];
}

// ---------------------------------------------------------------- layernorm
__global__ void layernorm_kernel(const float* __restrict__ in,
                                 const float* __restrict__ g,
                                 const float* __restrict__ bb,
                                 float* __restrict__ out) {
  const int E = 768;
  __shared__ float red[256];
  int row = blockIdx.x, tid = threadIdx.x;
  const float* x = in + (size_t)row * E;
  float s = 0.f;
  for (int i = tid; i < E; i += 256) s += x[i];
  red[tid] = s; __syncthreads();
  for (int off = 128; off > 0; off >>= 1) {
    if (tid < off) red[tid] += red[tid + off];
    __syncthreads();
  }
  float mean = red[0] * (1.0f / E);
  __syncthreads();
  float vs = 0.f;
  for (int i = tid; i < E; i += 256) { float d = x[i] - mean; vs += d * d; }
  red[tid] = vs; __syncthreads();
  for (int off = 128; off > 0; off >>= 1) {
    if (tid < off) red[tid] += red[tid + off];
    __syncthreads();
  }
  float rstd = rsqrtf(red[0] * (1.0f / E) + 1e-5f);
  float* o = out + (size_t)row * E;
  for (int i = tid; i < E; i += 256) o[i] = (x[i] - mean) * rstd * g[i] + bb[i];
}

// ---------------------------------------------------------------- weight repack
// fp32 weight [K,N] (bMode 1: QKV [H][K][64], col n = h*64+d) -> f16 tiles in
// WMMA-B-fragment order. Tile (n64, k32) = 3072 halves (6KB): element (k,n):
//   off = tile*3072 + ((n>>4)&3)*768 + (((k>>4)&1)*16 + (n&15))*24 + (k&15)
__global__ void repack_b_kernel(const float* __restrict__ Bw,
                                _Float16* __restrict__ out,
                                int K, int N, int bMode) {
  size_t idx = (size_t)blockIdx.x * 256 + threadIdx.x;
  if (idx >= (size_t)K * N) return;
  int k = (int)(idx / N), n = (int)(idx % N);
  size_t src = bMode ? ((size_t)(n >> 6) * K * 64 + (size_t)k * 64 + (n & 63))
                     : ((size_t)k * N + n);
  size_t tile = (size_t)(n >> 6) * (K >> 5) + (k >> 5);
  size_t off = tile * 3072 + (size_t)((n >> 4) & 3) * 32 * 24 +
               (size_t)(((k >> 4) & 1) * 16 + (n & 15)) * 24 + (k & 15);
  out[off] = (_Float16)Bw[src];
}

// ---------------------------------------------------------------- WMMA GEMM
// C[M,N] = epilogue(A[M,K](f32) * Bp(packed f16 tiles) + bias [+ resid])
__global__ void __launch_bounds__(128) gemm_kernel(
    const float* __restrict__ A, const _Float16* __restrict__ Bp,
    const float* __restrict__ bias, const float* __restrict__ resid,
    float* __restrict__ C, int M, int N, int K, int doRelu) {
  __shared__ _Float16 Ast[64][40];      // rows in natural k order (no permutation)
  __shared__ _Float16 Bst[2][3072];     // double-buffered async B tiles (6KB)
  int tid = threadIdx.x;
  int wid = tid >> 5, lane = tid & 31;
  int half = lane >> 4, l16 = lane & 15;
  int rowBase = blockIdx.y * 64;
  int colBase = blockIdx.x * 64;
  int nTiles = K >> 5;
  const _Float16* tile0 = Bp + (size_t)(colBase >> 6) * nTiles * 3072;

  v8f acc[4] = {};

  // prefetch first B tile (each thread async-copies 48 bytes)
  async_copy48((unsigned)(size_t)&Bst[0][tid * 24], tile0 + tid * 24);

  for (int kt = 0; kt < nTiles; ++kt) {
    int buf = kt & 1;
    bool more = (kt + 1) < nTiles;
    if (more)
      async_copy48((unsigned)(size_t)&Bst[buf ^ 1][tid * 24],
                   tile0 + (size_t)(kt + 1) * 3072 + tid * 24);

    // stage A tile 64x32: batched float2 loads, then f16 stores in natural k order
    int kb = kt * 32;
    float2 av[8];
#pragma unroll
    for (int it = 0; it < 8; ++it) {
      int p = tid + it * 128;
      int r = p >> 4, c = (p & 15) * 2;
      av[it] = *(const float2*)&A[(size_t)(rowBase + r) * K + (kb + c)];
    }
#pragma unroll
    for (int it = 0; it < 8; ++it) {
      int p = tid + it * 128;
      int r = p >> 4, c = (p & 15) * 2;
      int pp = c;
      Ast[r][pp]     = (_Float16)av[it].x;
      Ast[r][pp + 1] = (_Float16)av[it].y;
    }

    if (more) wait_async_keep3();   // current tile landed; prefetch in flight
    else      wait_async_all();
    __syncthreads();

    v16h a = frag_ld(&Ast[wid * 16 + l16][half * 16]);
#pragma unroll
    for (int c4 = 0; c4 < 4; ++c4) {
      v16h b = frag_ld(&Bst[buf][(c4 * 32 + lane) * 24]);
      acc[c4] = WMMA_F16(a, b, acc[c4]);
    }
    __syncthreads();
  }

  // epilogue (C layout: VGPR r -> row r+8*half, lane -> col)
#pragma unroll
  for (int c4 = 0; c4 < 4; ++c4) {
    int col = colBase + c4 * 16 + l16;
    float bval = bias ? bias[col] : 0.f;
#pragma unroll
    for (int r = 0; r < 8; ++r) {
      int row = rowBase + wid * 16 + r + 8 * half;
      float v = acc[c4][r] + bval;
      if (doRelu) v = fmaxf(v, 0.f);
      if (resid) v += resid[(size_t)row * N + col];
      C[(size_t)row * N + col] = v;
    }
  }
}

// ---------------------------------------------------------------- attention
// Flash-style causal attention, one wave per (b, h, 16-query tile).
// q,k,v,o are [B*T, 768] with head h at columns h*64..h*64+63.
__global__ void __launch_bounds__(128) attention_kernel(
    const float* __restrict__ qf, const float* __restrict__ kf,
    const float* __restrict__ vf, float* __restrict__ of) {
  const int T = 1024, HD = 64, E = 768;
  __shared__ _Float16 Pls[4][16][40];  // per-wave P in A-frag permuted order
  int wid = threadIdx.x >> 5;
  int lane = threadIdx.x & 31;
  int half = lane >> 4, l16 = lane & 15;
  int qBase = (blockIdx.x * 4 + wid) * 16;
  int h = blockIdx.y, b = blockIdx.z;
  const float scale = 0.03608439182435161f;  // 1/sqrt(768): ref scales by n_embed

  size_t batchOff = (size_t)b * T * E;
  size_t headOff = (size_t)h * HD;

  // Q fragments: vectorized (two 8-float runs per 32-wide K chunk)
  v16h qa[2];
  {
    const float* qrow = qf + batchOff + (size_t)(qBase + l16) * E + headOff;
#pragma unroll
    for (int ch = 0; ch < 2; ++ch) {
      float lo[8], hi[8];
      __builtin_memcpy(lo, qrow + ch * 32 + half * 16, 32);
      __builtin_memcpy(hi, qrow + ch * 32 + half * 16 + 8, 32);
#pragma unroll
      for (int i = 0; i < 8; ++i) {
        qa[ch][i]     = (_Float16)lo[i];
        qa[ch][i + 8] = (_Float16)hi[i];
      }
    }
  }

  v8f oacc[4] = {};
  float rmax[8], rsum[8];
#pragma unroll
  for (int r = 0; r < 8; ++r) { rmax[r] = -1e30f; rsum[r] = 0.f; }

  for (int sBase = 0; sBase <= qBase + 15; sBase += 32) {
    // S = Q * K^T for 32 keys (two 16-col tiles, 2 d-chunks each)
    v8f st[2] = {};
#pragma unroll
    for (int s2 = 0; s2 < 2; ++s2) {
      const float* krow = kf + batchOff + headOff +
                          (size_t)(sBase + s2 * 16 + l16) * E;
#pragma unroll
      for (int ch = 0; ch < 2; ++ch) {
        float fb[16];
        __builtin_memcpy(fb, krow + ch * 32 + half * 16, 64);
        v16h bk;
#pragma unroll
        for (int i = 0; i < 16; ++i) bk[i] = (_Float16)fb[i];
        st[s2] = WMMA_F16(qa[ch], bk, st[s2]);
      }
    }

    // online softmax update per row (rows live in C-layout halves)
    float corr[8];
#pragma unroll
    for (int r = 0; r < 8; ++r) {
      int t = qBase + r + 8 * half;
      float v0 = st[0][r] * scale;
      float v1 = st[1][r] * scale;
      if (sBase + l16 > t)      v0 = -1e30f;
      if (sBase + 16 + l16 > t) v1 = -1e30f;
      float mx = half16_max(fmaxf(v0, v1));
      float nm = fmaxf(rmax[r], mx);
      float c = __expf(rmax[r] - nm);
      float p0 = __expf(v0 - nm);
      float p1 = __expf(v1 - nm);
      rsum[r] = rsum[r] * c + half16_sum(p0 + p1);
      rmax[r] = nm;
      corr[r] = c;
      // store in A-frag permuted column order
      Pls[wid][r + 8 * half][aperm(l16)]      = (_Float16)p0;
      Pls[wid][r + 8 * half][aperm(16 + l16)] = (_Float16)p1;
    }
#pragma unroll
    for (int c4 = 0; c4 < 4; ++c4)
#pragma unroll
      for (int r = 0; r < 8; ++r) oacc[c4][r] *= corr[r];

    // P as A fragment: contiguous 32B per lane
    v16h pa = frag_ld(&Pls[wid][l16][half * 16]);

    // O += P * V
#pragma unroll
    for (int c4 = 0; c4 < 4; ++c4) {
      v16h bv;
      int d = c4 * 16 + l16;
#pragma unroll
      for (int i = 0; i < 16; ++i)
        bv[i] = (_Float16)vf[batchOff + headOff +
                             (size_t)(sBase + half * 8 + (i & 7) + (i >> 3) * 16) * E + d];
      oacc[c4] = WMMA_F16(pa, bv, oacc[c4]);
    }
  }

  // normalize and write heads-concatenated output
#pragma unroll
  for (int c4 = 0; c4 < 4; ++c4)
#pragma unroll
    for (int r = 0; r < 8; ++r) {
      int t = qBase + r + 8 * half;
      of[batchOff + (size_t)t * E + headOff + c4 * 16 + l16] =
          oacc[c4][r] / rsum[r];
    }
}

// ---------------------------------------------------------------- loss
__global__ void zero_kernel(float* p) {
  if (threadIdx.x == 0 && blockIdx.x == 0) *p = 0.f;
}

__global__ void loss_kernel(const float* __restrict__ logits,
                            const int* __restrict__ targets,
                            float* __restrict__ loss, int Vn, float invM) {
  __shared__ float red[256];
  int row = blockIdx.x, tid = threadIdx.x;
  const float* lg = logits + (size_t)row * Vn;
  float mx = -1e30f;
  for (int i = tid; i < Vn; i += 256) mx = fmaxf(mx, lg[i]);
  red[tid] = mx; __syncthreads();
  for (int off = 128; off > 0; off >>= 1) {
    if (tid < off) red[tid] = fmaxf(red[tid], red[tid + off]);
    __syncthreads();
  }
  float m = red[0];
  __syncthreads();
  float s = 0.f;
  for (int i = tid; i < Vn; i += 256) s += __expf(lg[i] - m);
  red[tid] = s; __syncthreads();
  for (int off = 128; off > 0; off >>= 1) {
    if (tid < off) red[tid] += red[tid + off];
    __syncthreads();
  }
  if (tid == 0) {
    float lt = lg[targets[row]];
    atomicAdd(loss, -(lt - m - logf(red[0])) * invM);
  }
}

// ---------------------------------------------------------------- driver
extern "C" void kernel_launch(void* const* d_in, const int* in_sizes, int n_in,
                              void* d_out, int out_size, void* d_ws, size_t ws_size,
                              hipStream_t stream) {
  (void)in_sizes; (void)n_in; (void)out_size; (void)ws_size;
  const int Bb = 4, Tt = 1024, E = 768, L = 6, Vv = 32000, F = 3072;
  const int M = Bb * Tt;  // 4096 rows

  const int*   token_idx = (const int*)d_in[0];
  const int*   targets   = (const int*)d_in[1];
  const float* tok_emb   = (const float*)d_in[2];
  const float* pos_emb   = (const float*)d_in[3];
  const float* Wq        = (const float*)d_in[4];
  const float* Wk        = (const float*)d_in[5];
  const float* Wv        = (const float*)d_in[6];
  const float* Wp        = (const float*)d_in[7];
  const float* bp        = (const float*)d_in[8];
  const float* ln1_g     = (const float*)d_in[9];
  const float* ln1_b     = (const float*)d_in[10];
  const float* ln2_g     = (const float*)d_in[11];
  const float* ln2_b     = (const float*)d_in[12];
  const float* W1        = (const float*)d_in[13];
  const float* b1        = (const float*)d_in[14];
  const float* W2        = (const float*)d_in[15];
  const float* b2        = (const float*)d_in[16];
  const float* lnf_g     = (const float*)d_in[17];
  const float* lnf_b     = (const float*)d_in[18];
  const float* lnf2_g    = (const float*)d_in[19];
  const float* lnf2_b    = (const float*)d_in[20];
  const float* Wout      = (const float*)d_in[21];
  const float* bout      = (const float*)d_in[22];

  float* ws = (float*)d_ws;
  size_t SZ = (size_t)M * E;
  float* x  = ws;            // [M,E] activation stream
  float* h  = ws + 1 * SZ;   // post-LN1
  float* q  = ws + 2 * SZ;
  float* k  = ws + 3 * SZ;
  float* v  = ws + 4 * SZ;
  float* o  = ws + 5 * SZ;   // attention out (heads concatenated)
  float* hp = ws + 6 * SZ;   // h + o@Wp + bp
  float* h2 = ws + 7 * SZ;   // post-LN2
  float* ff = ws + 8 * SZ;   // [M,4E] relu MLP hidden (uses 4*SZ)

  // f16 packed-weight arena (reused per layer; lm head uses it alone)
  _Float16* wf = (_Float16*)(ws + 12 * SZ);
  auto packed = [](int K, int N) { return (size_t)(N / 64) * (K / 32) * 3072; };
  _Float16* wq16 = wf;
  _Float16* wk16 = wq16 + packed(E, E);
  _Float16* wv16 = wk16 + packed(E, E);
  _Float16* wp16 = wv16 + packed(E, E);
  _Float16* w116 = wp16 + packed(E, E);
  _Float16* w216 = w116 + packed(E, F);

  float* logits = (float*)d_out;
  float* loss = logits + (size_t)M * Vv;

  size_t total = (size_t)M * E;
  embed_kernel<<<(int)(total / 256), 256, 0, stream>>>(token_idx, tok_emb,
                                                       pos_emb, x, total);

  dim3 gE(E / 64, M / 64), gF(F / 64, M / 64), gV(Vv / 64, M / 64);
  int rEE = E * E / 256, rEF = E * F / 256;
  for (int l = 0; l < L; ++l) {
    size_t wOff = (size_t)l * E * E;    // H*E*HD == E*E
    repack_b_kernel<<<rEE, 256, 0, stream>>>(Wq + wOff, wq16, E, E, 1);
    repack_b_kernel<<<rEE, 256, 0, stream>>>(Wk + wOff, wk16, E, E, 1);
    repack_b_kernel<<<rEE, 256, 0, stream>>>(Wv + wOff, wv16, E, E, 1);
    repack_b_kernel<<<rEE, 256, 0, stream>>>(Wp + wOff, wp16, E, E, 0);
    repack_b_kernel<<<rEF, 256, 0, stream>>>(W1 + (size_t)l * E * F, w116, E, F, 0);
    repack_b_kernel<<<rEF, 256, 0, stream>>>(W2 + (size_t)l * F * E, w216, F, E, 0);

    layernorm_kernel<<<M, 256, 0, stream>>>(x, ln1_g + l * E, ln1_b + l * E, h);
    gemm_kernel<<<gE, 128, 0, stream>>>(h, wq16, nullptr, nullptr, q, M, E, E, 0);
    gemm_kernel<<<gE, 128, 0, stream>>>(h, wk16, nullptr, nullptr, k, M, E, E, 0);
    gemm_kernel<<<gE, 128, 0, stream>>>(h, wv16, nullptr, nullptr, v, M, E, E, 0);
    attention_kernel<<<dim3(Tt / 64, 12, Bb), 128, 0, stream>>>(q, k, v, o);
    gemm_kernel<<<gE, 128, 0, stream>>>(o, wp16, bp + l * E, h, hp, M, E, E, 0);
    layernorm_kernel<<<M, 256, 0, stream>>>(hp, ln2_g + l * E, ln2_b + l * E, h2);
    gemm_kernel<<<gF, 128, 0, stream>>>(h2, w116, b1 + (size_t)l * F, nullptr,
                                        ff, M, F, E, 1);
    gemm_kernel<<<gE, 128, 0, stream>>>(ff, w216, b2 + l * E, h2, x, M, E, F, 0);
  }

  layernorm_kernel<<<M, 256, 0, stream>>>(x, lnf_g, lnf_b, h);
  layernorm_kernel<<<M, 256, 0, stream>>>(h, lnf2_g, lnf2_b, h2);
  repack_b_kernel<<<E * Vv / 256, 256, 0, stream>>>(Wout, wf, E, Vv, 0);
  gemm_kernel<<<gV, 128, 0, stream>>>(h2, wf, bout, nullptr, logits, M, Vv, E, 0);
  zero_kernel<<<1, 1, 0, stream>>>(loss);
  loss_kernel<<<M, 256, 0, stream>>>(logits, targets, loss, Vv, 1.0f / M);
}
